// ConvBnSign_8237747274235
// MI455X (gfx1250) — matrix operand order repaired and checked by reading, not compiled, not run on hardware
//
#include <hip/hip_runtime.h>

typedef __attribute__((ext_vector_type(16))) _Float16 v16h;
typedef __attribute__((ext_vector_type(8)))  _Float16 v8h;
typedef __attribute__((ext_vector_type(8)))  float    v8f;
typedef __attribute__((ext_vector_type(4)))  int      v4i;
typedef __attribute__((address_space(1)))    v4i      gv4i;   // global int4
typedef __attribute__((address_space(3)))    v4i      lv4i;   // LDS int4

#define N_    32
#define C_    128
#define H_    56
#define W_    56
#define O_    256
#define K_    1152          // 9 taps * 128 in-channels
#define HW_   (H_*W_)       // 3136
#define M_    (N_*HW_)      // 100352 output pixels
#define NKB   (K_/32)       // 36 (32-wide K blocks, WMMA granularity)
#define BM    32            // pixels per block
#define LDSA_ROW 72         // halfs per A row (144 B = 9*16 B -> conflict-free b128)
#define PH    58            // padded spatial dim
#define XP_HALFS ((size_t)N_*PH*PH*C_)   // 13,778,944 halfs

#if __has_builtin(__builtin_amdgcn_global_load_async_to_lds_b128)
#define HAVE_ASYNC_LDS 1
#endif

__device__ __forceinline__ void wait_asynccnt0() {
#if __has_builtin(__builtin_amdgcn_s_wait_asynccnt)
    __builtin_amdgcn_s_wait_asynccnt(0);
#else
    asm volatile("s_wait_asynccnt 0" ::: "memory");
#endif
}

// ---------------------------------------------------------------------------
// Kernel 0a: zero the padded-NHWC f16 activation buffer (borders = conv pad)
// ---------------------------------------------------------------------------
__global__ __launch_bounds__(256)
void zero_xpad(uint4* __restrict__ xp16) {
    const size_t i = (size_t)blockIdx.x * blockDim.x + threadIdx.x;
    xp16[i] = make_uint4(0u, 0u, 0u, 0u);
}

// ---------------------------------------------------------------------------
// Kernel 0b: transpose x NCHW f32 -> padded NHWC f16: xp[n][h+1][w+1][c]
// One block per (n,h) row; LDS tile for the c<->w transpose.
// ---------------------------------------------------------------------------
__global__ __launch_bounds__(256)
void prep_x(const float* __restrict__ x, _Float16* __restrict__ xp) {
    const int nh = blockIdx.x;             // n*56 + h
    const int n  = nh / H_;
    const int h  = nh - n * H_;
    __shared__ _Float16 tile[C_ * 57];     // [c][w], stride 57 (bank spread)
    const int t = threadIdx.x;
    for (int idx = t; idx < C_ * W_; idx += 256) {      // coalesced along w
        const int c = idx / W_;
        const int w = idx - c * W_;
        tile[c * 57 + w] = (_Float16)x[(((size_t)n * C_ + c) * H_ + h) * W_ + w];
    }
    __syncthreads();
    for (int idx = t; idx < W_ * (C_ / 8); idx += 256) { // vector stores along c
        const int w  = idx >> 4;            // C_/8 == 16
        const int c8 = (idx & 15) << 3;
        v8h v;
#pragma unroll
        for (int i = 0; i < 8; ++i) v[i] = tile[(c8 + i) * 57 + w];
        *(v8h*)&xp[(((size_t)n * PH + (h + 1)) * PH + (w + 1)) * C_ + c8] = v;
    }
}

// ---------------------------------------------------------------------------
// Kernel 1: alpha = mean|w| per out-channel; pack sign(w)*alpha as f16 into
// B-fragment layout bpack[kb][o][kl].  K index = tap*128 + c, tap = kh*3+kw.
// ---------------------------------------------------------------------------
__global__ __launch_bounds__(128)
void prep_pack_weights(const float* __restrict__ w, _Float16* __restrict__ bpack) {
    const int o = blockIdx.x;
    const int t = threadIdx.x;
    __shared__ float red[128];
    float s = 0.f;
    for (int i = t; i < K_; i += 128) s += fabsf(w[o * K_ + i]);
    red[t] = s;
    __syncthreads();
    for (int off = 64; off > 0; off >>= 1) {
        if (t < off) red[t] += red[t + off];
        __syncthreads();
    }
    const float alpha = red[0] * (1.0f / (float)K_);
    for (int k = t; k < K_; k += 128) {
        const int c = k & 127;
        const int p = k >> 7;                       // tap 0..8
        const float wv = w[(o * C_ + c) * 9 + p];   // OIHW flat
        const float bv = (wv >= 0.f) ? alpha : -alpha;
        const int kb = k >> 5, kl = k & 31;
        bpack[((size_t)(kb * O_ + o) << 5) + kl] = (_Float16)bv;
    }
}

__global__ void zero_stats(float* stats) {
    stats[blockIdx.x * blockDim.x + threadIdx.x] = 0.f;
}

// ---------------------------------------------------------------------------
// Kernel 2: implicit-GEMM conv via v_wmma_f32_16x16x32_f16.
// Block = 256 thr (8 wave32). Tile: 32 pixels x 256 channels.
// K-step 64 (one tap per 2 steps); LDS double-buffered -> 1 barrier/step;
// A staged via GLOBAL_LOAD_ASYNC_TO_LDS_B128 (ASYNCcnt) or b128 load/store.
// ---------------------------------------------------------------------------
__global__ __launch_bounds__(256)
void conv_wmma(const _Float16* __restrict__ xp, const _Float16* __restrict__ bpack,
               float* __restrict__ y, float* __restrict__ sum,
               float* __restrict__ sumsq) {
    __shared__ _Float16 aLds[2][BM * LDSA_ROW];
    __shared__ int pixoff[BM];                 // n*O_*HW_ + h*56 + w

    const int t    = threadIdx.x;
    const int lane = t & 31;
    const int wv   = t >> 5;                   // wave 0..7
    const int col  = lane & 15;
    const int half = lane >> 4;                // 0/1
    const int n0   = wv * 32;                  // channel base for this wave
    const int m0   = blockIdx.x * BM;

    if (t < BM) {                              // y-store offsets
        const int gm2 = m0 + t;
        const int n2  = gm2 / HW_;
        pixoff[t] = n2 * (O_ * HW_) + (gm2 - n2 * HW_);
    }

    // staging: 8 threads/pixel, each thread = 8 consecutive channels (16 B)
    const int sm  = t >> 3;                    // pixel 0..31
    const int cb8 = (t & 7) << 3;              // 0,8,..,56
    const int gm  = m0 + sm;
    const int n   = gm / HW_;
    const int rem = gm - n * HW_;
    const int h   = rem / W_;
    const int wpx = rem - h * W_;

    // A-fragment layout: contiguous k (8-groups in natural order) in LDS column
    const int blk = cb8 >> 5;
    const int g   = (cb8 & 31) >> 3;
    const int jg  = g;
    const int aOff = sm * LDSA_ROW + blk * 32 + jg * 8;

    // padded-NHWC base: tap p reads padded (h + p/3, w + p%3)
    const _Float16* xb =
        xp + (((size_t)n * PH + h) * PH + wpx) * C_ + cb8;
    auto gsrc = [&](int s) -> const _Float16* {
        const int p = s >> 1;
        return xb + (((p / 3) * PH + (p % 3)) * C_) + (s & 1) * 64;
    };

    v8f acc00 = {}, acc01 = {}, acc10 = {}, acc11 = {};

    // -------- software pipeline: prologue stages step 0 into buffer 0 ------
#ifdef HAVE_ASYNC_LDS
    __builtin_amdgcn_global_load_async_to_lds_b128(
        (gv4i*)gsrc(0), (lv4i*)&aLds[0][aOff], 0, 0);
#else
    {
        const v8h av0 = *(const v8h*)gsrc(0);
        *(v8h*)&aLds[0][aOff] = av0;
    }
#endif

    for (int s = 0; s < 18; ++s) {
        const int cur = s & 1;
#ifdef HAVE_ASYNC_LDS
        wait_asynccnt0();                      // pending copy into buf[cur] done
#endif
        __syncthreads();                       // publish buf[cur]; buf[!cur] free

#ifdef HAVE_ASYNC_LDS
        if (s + 1 < 18)
            __builtin_amdgcn_global_load_async_to_lds_b128(
                (gv4i*)gsrc(s + 1), (lv4i*)&aLds[cur ^ 1][aOff], 0, 0);
#else
        v8h av;
        if (s + 1 < 18) av = *(const v8h*)gsrc(s + 1);
#endif
        // prefetch packed-B two steps (four K-blocks) ahead (global_prefetch_b8)
        if (2 * s + 4 < NKB)
            __builtin_prefetch(bpack + (((size_t)(2 * s + 4) * O_) << 5) +
                               ((size_t)t << 6), 0, 1);

        // ---- compute: 2 M-subtiles x 2 N-subtiles x 2 K-halves = 8 WMMA ----
#pragma unroll
        for (int kt = 0; kt < 2; ++kt) {
            const int kb = 2 * s + kt;
            const _Float16* a0p = &aLds[cur][col * LDSA_ROW + kt * 32 + half * 16];
            const _Float16* a1p = a0p + 16 * LDSA_ROW;
            const v16h a0 = __builtin_shufflevector(
                *(const v8h*)a0p, *(const v8h*)(a0p + 8),
                0,1,2,3,4,5,6,7,8,9,10,11,12,13,14,15);
            const v16h a1 = __builtin_shufflevector(
                *(const v8h*)a1p, *(const v8h*)(a1p + 8),
                0,1,2,3,4,5,6,7,8,9,10,11,12,13,14,15);
            const v16h b0 = *(const v16h*)(bpack +
                (((size_t)kb * O_ + (n0 + col)) << 5) + half * 16);
            const v16h b1 = *(const v16h*)(bpack +
                (((size_t)kb * O_ + (n0 + 16 + col)) << 5) + half * 16);

            acc00 = __builtin_amdgcn_wmma_f32_16x16x32_f16(
                        false, a0, false, b0, (short)0, acc00, false, false);
            acc01 = __builtin_amdgcn_wmma_f32_16x16x32_f16(
                        false, a0, false, b1, (short)0, acc01, false, false);
            acc10 = __builtin_amdgcn_wmma_f32_16x16x32_f16(
                        false, a1, false, b0, (short)0, acc10, false, false);
            acc11 = __builtin_amdgcn_wmma_f32_16x16x32_f16(
                        false, a1, false, b1, (short)0, acc11, false, false);
        }

#ifndef HAVE_ASYNC_LDS
        if (s + 1 < 18) *(v8h*)&aLds[cur ^ 1][aOff] = av;
#endif
    }

    // ---- per-channel batch-stat partials (C/D layout: lane = col N,
    //      rows M = mt*16 + half*8 + r); reduce across half-waves ----
    float s0 = 0.f, q0 = 0.f, s1 = 0.f, q1 = 0.f;
#pragma unroll
    for (int r = 0; r < 8; ++r) {
        s0 += acc00[r] + acc10[r];
        q0 += acc00[r] * acc00[r] + acc10[r] * acc10[r];
        s1 += acc01[r] + acc11[r];
        q1 += acc01[r] * acc01[r] + acc11[r] * acc11[r];
    }
    s0 += __shfl_xor(s0, 16, 32); q0 += __shfl_xor(q0, 16, 32);
    s1 += __shfl_xor(s1, 16, 32); q1 += __shfl_xor(q1, 16, 32);
    if (half == 0) {
        atomicAdd(&sum[n0 + col], s0);        atomicAdd(&sumsq[n0 + col], q0);
        atomicAdd(&sum[n0 + 16 + col], s1);   atomicAdd(&sumsq[n0 + 16 + col], q1);
    }

    // ---- store y (NCHW f32) ----
    const size_t ch0 = (size_t)(n0 + col) * HW_;
    const size_t ch1 = (size_t)(n0 + 16 + col) * HW_;
#pragma unroll
    for (int r = 0; r < 8; ++r) {
        const int po0 = pixoff[half * 8 + r];         // M-subtile 0
        const int po1 = pixoff[16 + half * 8 + r];    // M-subtile 1
        y[(size_t)po0 + ch0] = acc00[r];
        y[(size_t)po0 + ch1] = acc01[r];
        y[(size_t)po1 + ch0] = acc10[r];
        y[(size_t)po1 + ch1] = acc11[r];
    }
}

// ---------------------------------------------------------------------------
// Kernel 3: fold BN into per-channel scale/shift
// ---------------------------------------------------------------------------
__global__ __launch_bounds__(256)
void finalize_stats(const float* __restrict__ sum, const float* __restrict__ sumsq,
                    const float* __restrict__ gamma, const float* __restrict__ beta,
                    float* __restrict__ scale, float* __restrict__ shift) {
    const int c = threadIdx.x;
    const float inv_m = 1.0f / (float)M_;
    const float mean  = sum[c] * inv_m;
    const float var   = sumsq[c] * inv_m - mean * mean;
    const float inv   = rsqrtf(var + 1e-5f);
    const float sc    = gamma[c] * inv;
    scale[c] = sc;
    shift[c] = beta[c] - mean * sc;
}

// ---------------------------------------------------------------------------
// Kernel 4: out = sign(y*scale + shift), float4 vectorized (pure bandwidth)
// ---------------------------------------------------------------------------
__global__ __launch_bounds__(256)
void bn_sign(const float* __restrict__ y, const float* __restrict__ scale,
             const float* __restrict__ shift, float* __restrict__ out) {
    const size_t i4 = (size_t)blockIdx.x * blockDim.x + threadIdx.x;
    const int ch = (int)((i4 / (HW_ / 4)) & 255);
    const float sc = scale[ch], sh = shift[ch];
    const float4 v = ((const float4*)y)[i4];
    float4 r;
    r.x = (v.x * sc + sh) >= 0.f ? 1.f : -1.f;
    r.y = (v.y * sc + sh) >= 0.f ? 1.f : -1.f;
    r.z = (v.z * sc + sh) >= 0.f ? 1.f : -1.f;
    r.w = (v.w * sc + sh) >= 0.f ? 1.f : -1.f;
    ((float4*)out)[i4] = r;
}

// ---------------------------------------------------------------------------
extern "C" void kernel_launch(void* const* d_in, const int* in_sizes, int n_in,
                              void* d_out, int out_size, void* d_ws, size_t ws_size,
                              hipStream_t stream) {
    (void)in_sizes; (void)n_in; (void)out_size; (void)ws_size;
    const float* x      = (const float*)d_in[0];
    const float* weight = (const float*)d_in[1];
    const float* gamma  = (const float*)d_in[2];
    const float* beta   = (const float*)d_in[3];
    float*       out    = (float*)d_out;

    float* ws    = (float*)d_ws;
    float* y     = ws;                          // 25,690,112 f32
    float* sum   = ws + (size_t)M_ * O_;        // 256
    float* sumsq = sum + 256;                   // 256
    float* scale = sum + 512;                   // 256
    float* shift = sum + 768;                   // 256
    _Float16* bpack = (_Float16*)(sum + 1024);  // 36*256*32 halfs
    _Float16* xpad  = (_Float16*)(sum + 1024 + (NKB * O_ * 32) / 2); // 16B aligned

    zero_xpad<<<(int)(XP_HALFS * 2 / 16 / 256), 256, 0, stream>>>((uint4*)xpad);
    prep_x<<<N_ * H_, 256, 0, stream>>>(x, xpad);
    prep_pack_weights<<<O_, 128, 0, stream>>>(weight, bpack);
    zero_stats<<<1, 1024, 0, stream>>>(sum);
    conv_wmma<<<M_ / BM, 256, 0, stream>>>(xpad, bpack, y, sum, sumsq);
    finalize_stats<<<1, 256, 0, stream>>>(sum, sumsq, gamma, beta, scale, shift);
    bn_sign<<<(M_ * O_ / 4) / 256, 256, 0, stream>>>(y, scale, shift, out);
}
